// ACGCNN_68341519614785
// MI455X (gfx1250) — compile-verified, matrix operand order checked by reading, not run on hardware
//
#include <hip/hip_runtime.h>
#include <hip/hip_bf16.h>
#include <math.h>

// ---------------------------------------------------------------------------
// ALIGNN forward for MI455X (gfx1250).
// Dense GEMMs: v_wmma_f32_16x16x32_bf16 with pre-padded bf16 activations and
// fragment-packed bf16 weights (tile count padded to 4) -> branch-free K loop:
// 2 A b128 loads + 8 B b128 loads + 4 WMMAs per iteration.
// Graph gather/scatter/BN: fp32 streams (memory-bound @ 23.3 TB/s HBM).
// ---------------------------------------------------------------------------

typedef __attribute__((ext_vector_type(16))) __bf16 v16bf;
typedef __attribute__((ext_vector_type(8)))  __bf16 v8bf;
typedef __attribute__((ext_vector_type(8)))  float  v8f;

#define NN_  20000
#define NE_  320000
#define NA_  640000
#define NG_  256
#define NFD_ 92
#define EFD_ 40
#define FCD_ 256
#define LAYERS_ 3

__device__ __forceinline__ float softplus_f(float x) {
  return fmaxf(x, 0.f) + log1pf(expf(-fabsf(x)));
}
__device__ __forceinline__ float sigmoid_f(float x) {
  return 1.f / (1.f + expf(-x));
}
__device__ __forceinline__ unsigned short bf16_bits(float x) {
  __bf16 b = (__bf16)x;
  return __builtin_bit_cast(unsigned short, b);
}

// ---------------------------------------------------------------------------
// Activation convert: X fp32 [M,K] -> Y bf16 [M,Kp], zero-padded (Kp % 32 == 0)
// ---------------------------------------------------------------------------
__global__ void to_bf16_pad(const float* __restrict__ X,
                            unsigned short* __restrict__ Y,
                            long M, int K, int Kp)
{
  long idx = (long)blockIdx.x * blockDim.x + threadIdx.x;
  if (idx >= M * Kp) return;
  long m = idx / Kp;
  int  k = (int)(idx - m * Kp);
  float v = (k < K) ? X[m * K + k] : 0.f;
  Y[idx] = bf16_bits(v);
}

// ---------------------------------------------------------------------------
// Weight pack: B fp32 [K,N] -> Bp bf16 fragments [kchunk][ntile4][lane][16]
// Fragment element order matches the CDNA5 16-bit B layout:
//   lanes 0-15 : col = lane,    K pairs {0,2,4,6,16,18,20,22}+kb
//   lanes 16-31: col = lane-16, K pairs {8,10,12,14,24,26,28,30}+kb
// Zero-pads K, N AND the tile count (to a multiple of 4), so the GEMM K-loop
// needs no bounds checks at all.
// ---------------------------------------------------------------------------
__global__ void pack_B_frag(const float* __restrict__ B,
                            unsigned short* __restrict__ Bp,
                            int K, int N, int ntiles4)
{
  const int lane = threadIdx.x;             // 32 threads
  const int nt   = blockIdx.x % ntiles4;
  const int kc   = blockIdx.x / ntiles4;
  const int khalf8 = (lane >> 4) ? 8 : 0;
  const int col  = nt * 16 + (lane & 15);
  unsigned short* out = Bp + ((size_t)blockIdx.x * 32 + lane) * 16;
#pragma unroll
  for (int i = 0; i < 8; ++i) {
    const int k = kc * 32 + 2 * (i & 3) + ((i & 4) ? 16 : 0) + khalf8;
    float b0 = 0.f, b1 = 0.f;
    if (col < N) {
      if (k < K)     b0 = B[(size_t)k * N + col];
      if (k + 1 < K) b1 = B[(size_t)(k + 1) * N + col];
    }
    out[2 * i]     = bf16_bits(b0);
    out[2 * i + 1] = bf16_bits(b1);
  }
}

// ---------------------------------------------------------------------------
// WMMA GEMM: C[M,N] = Abf[M,Kp] @ B + bias[N]
// One wave -> 16 rows x 4 N-tiles; A fragment reused across the 4 WMMAs.
// Branch-free inner loop (tile count pre-padded to multiple of 4).
// ---------------------------------------------------------------------------
__global__ void __launch_bounds__(128)
gemm_wmma(const unsigned short* __restrict__ Abf,
          const unsigned short* __restrict__ Bp,
          const float* __restrict__ bias,
          float* __restrict__ C,
          int N, int Kp, int ntiles4, int mtiles)
{
  const int lane  = threadIdx.x & 31;
  const int wave  = threadIdx.x >> 5;
  const int tileM = blockIdx.y * 4 + wave;
  if (tileM >= mtiles) return;
  const int nt0 = blockIdx.x * 4;

  const int mrow   = tileM * 16 + (lane & 15);
  const int khalf8 = (lane >> 4) ? 8 : 0;
  const unsigned short* arow = Abf + (size_t)mrow * Kp;

  v8f acc[4] = { {}, {}, {}, {} };
  const int kchunks = Kp >> 5;

  for (int kc = 0; kc < kchunks; ++kc) {
    union { v16bf v; v8bf h[2]; } a;
    a.h[0] = *(const v8bf*)(arow + kc * 32 + khalf8);
    a.h[1] = *(const v8bf*)(arow + kc * 32 + 16 + khalf8);
    const unsigned short* bbase =
        Bp + (((size_t)kc * ntiles4 + nt0) * 32 + lane) * 16;
    v16bf b0 = *(const v16bf*)(bbase + 0 * 32 * 16);
    v16bf b1 = *(const v16bf*)(bbase + 1 * 32 * 16);
    v16bf b2 = *(const v16bf*)(bbase + 2 * 32 * 16);
    v16bf b3 = *(const v16bf*)(bbase + 3 * 32 * 16);
    acc[0] = __builtin_amdgcn_wmma_f32_16x16x32_bf16(false, a.v, false, b0,
                                                     (short)0, acc[0], false, false);
    acc[1] = __builtin_amdgcn_wmma_f32_16x16x32_bf16(false, a.v, false, b1,
                                                     (short)0, acc[1], false, false);
    acc[2] = __builtin_amdgcn_wmma_f32_16x16x32_bf16(false, a.v, false, b2,
                                                     (short)0, acc[2], false, false);
    acc[3] = __builtin_amdgcn_wmma_f32_16x16x32_bf16(false, a.v, false, b3,
                                                     (short)0, acc[3], false, false);
  }

  const int rowbase = tileM * 16 + ((lane >> 4) ? 8 : 0);
#pragma unroll
  for (int t = 0; t < 4; ++t) {
    const int ncol = (nt0 + t) * 16 + (lane & 15);
    if (ncol < N) {                       // also excludes padded tiles
      const float bv = bias ? bias[ncol] : 0.f;
#pragma unroll
      for (int r = 0; r < 8; ++r) {
        C[(size_t)(rowbase + r) * N + ncol] = acc[t][r] + bv;
      }
    }
  }
}

// ---------------------------------------------------------------------------
// Elementwise / featurization kernels
// ---------------------------------------------------------------------------
__global__ void zero_kernel(float* __restrict__ p, long n) {
  long i = (long)blockIdx.x * blockDim.x + threadIdx.x;
  if (i < n) p[i] = 0.f;
}

__global__ void bondlen_kernel(const float* __restrict__ r, float* __restrict__ bl, int n) {
  int i = blockIdx.x * blockDim.x + threadIdx.x;
  if (i >= n) return;
  float x = r[3 * i], y = r[3 * i + 1], z = r[3 * i + 2];
  bl[i] = sqrtf(x * x + y * y + z * z);
}

__global__ void rbf_kernel(const float* __restrict__ x, float* __restrict__ out,
                           long n, float vmin, float step, int bins) {
  long idx = (long)blockIdx.x * blockDim.x + threadIdx.x;
  if (idx >= n * bins) return;
  long i = idx / bins;
  int  b = (int)(idx - i * bins);
  float gamma = 1.f / step;
  float d = x[i] - (vmin + step * b);
  out[idx] = expf(-gamma * d * d);
}

// ---------------------------------------------------------------------------
// Fused gather + message + column stats (thread = fixed column):
//   m[e,c] = Me[e,c] + Hsrc[src[e],c] + Hdst[dst[e],c]  (in-place into Mbuf)
// ---------------------------------------------------------------------------
__global__ void gather_msg_stats(const float* __restrict__ Hsrc,
                                 const float* __restrict__ Hdst,
                                 float* __restrict__ Mbuf,
                                 const int* __restrict__ src,
                                 const int* __restrict__ dst,
                                 float* __restrict__ colsum,
                                 float* __restrict__ colsumsq,
                                 int Ne, int Fo, int edgesPerBlock)
{
  const int c = threadIdx.x;
  if (c >= Fo) return;
  int e0 = blockIdx.x * edgesPerBlock;
  int e1 = e0 + edgesPerBlock;
  if (e1 > Ne) e1 = Ne;
  float s = 0.f, ss = 0.f;
  for (int e = e0; e < e1; ++e) {
    __builtin_prefetch(&Mbuf[(long)(e + 1) * Fo + c], 0, 1);
    const int si = src[e], di = dst[e];
    float m = Mbuf[(long)e * Fo + c]
            + Hsrc[(long)si * Fo + c]
            + Hdst[(long)di * Fo + c];
    Mbuf[(long)e * Fo + c] = m;
    s += m;
    ss += m * m;
  }
  atomicAdd(&colsum[c], s);
  atomicAdd(&colsumsq[c], ss);
}

__global__ void col_stats(const float* __restrict__ X,
                          float* __restrict__ colsum,
                          float* __restrict__ colsumsq,
                          int Nrows, int F, int rowsPerBlock)
{
  const int c = threadIdx.x;
  if (c >= F) return;
  int r0 = blockIdx.x * rowsPerBlock;
  int r1 = r0 + rowsPerBlock;
  if (r1 > Nrows) r1 = Nrows;
  float s = 0.f, ss = 0.f;
  for (int r = r0; r < r1; ++r) {
    float v = X[(long)r * F + c];
    s += v;
    ss += v * v;
  }
  atomicAdd(&colsum[c], s);
  atomicAdd(&colsumsq[c], ss);
}

__global__ void bn_finalize(const float* __restrict__ colsum,
                            const float* __restrict__ colsumsq,
                            const float* __restrict__ g,
                            const float* __restrict__ b,
                            float* __restrict__ scale,
                            float* __restrict__ shift,
                            int F, float invN)
{
  int c = blockIdx.x * blockDim.x + threadIdx.x;
  if (c >= F) return;
  float mu  = colsum[c] * invN;
  float var = colsumsq[c] * invN - mu * mu;
  float sc  = g[c] * rsqrtf(var + 1e-5f);
  scale[c] = sc;
  shift[c] = b[c] - mu * sc;
}

__global__ void gate_scatter(const float* __restrict__ Mbuf,
                             const float* __restrict__ scale,
                             const float* __restrict__ shift,
                             const int* __restrict__ dst,
                             float* __restrict__ Hacc,
                             long total, int Fin)
{
  long idx = (long)blockIdx.x * blockDim.x + threadIdx.x;
  if (idx >= total) return;
  long e = idx / Fin;
  int  c = (int)(idx - e * Fin);
  const int Fo = 2 * Fin;
  float mf = Mbuf[e * Fo + c]       * scale[c]       + shift[c];
  float ms = Mbuf[e * Fo + c + Fin] * scale[c + Fin] + shift[c + Fin];
  float v  = sigmoid_f(mf) * softplus_f(ms);
  atomicAdd(&Hacc[(long)dst[e] * Fin + c], v);
}

__global__ void node_update(float* __restrict__ nf,
                            const float* __restrict__ Hacc,
                            const float* __restrict__ scale,
                            const float* __restrict__ shift,
                            long total, int F)
{
  long idx = (long)blockIdx.x * blockDim.x + threadIdx.x;
  if (idx >= total) return;
  int c = (int)(idx % F);
  float h = Hacc[idx] * scale[c] + shift[c];
  nf[idx] = softplus_f(nf[idx] + h);
}

// ---------------------------------------------------------------------------
// Readout
// ---------------------------------------------------------------------------
__global__ void pool_kernel(const float* __restrict__ nf,
                            const int* __restrict__ gid,
                            float* __restrict__ sums,
                            float* __restrict__ cnts,
                            long total, int F)
{
  long idx = (long)blockIdx.x * blockDim.x + threadIdx.x;
  if (idx >= total) return;
  long i = idx / F;
  int  c = (int)(idx - i * F);
  int  g = gid[i];
  atomicAdd(&sums[(long)g * F + c], nf[idx]);
  if (c == 0) atomicAdd(&cnts[g], 1.f);
}

__global__ void pool_finalize(float* __restrict__ feats,
                              const float* __restrict__ sums,
                              const float* __restrict__ cnts,
                              long total, int F)
{
  long idx = (long)blockIdx.x * blockDim.x + threadIdx.x;
  if (idx >= total) return;
  long g = idx / F;
  float cnt = fmaxf(cnts[g], 1.f);
  feats[idx] = softplus_f(sums[idx] / cnt);
}

__global__ void softplus2_kernel(float* __restrict__ x, long n) {
  long i = (long)blockIdx.x * blockDim.x + threadIdx.x;
  if (i >= n) return;
  x[i] = softplus_f(softplus_f(x[i]));
}

__global__ void out_kernel(const float* __restrict__ feats,
                           const float* __restrict__ outW,
                           const float* __restrict__ outb,
                           float* __restrict__ out, int G, int F)
{
  int g = blockIdx.x * blockDim.x + threadIdx.x;
  if (g >= G) return;
  float s = outb[0];
  for (int k = 0; k < F; ++k) s += feats[(long)g * F + k] * outW[k];
  out[g] = s;
}

// ---------------------------------------------------------------------------
// Host-side orchestration
// ---------------------------------------------------------------------------
static inline void launch_zero(float* p, long n, hipStream_t s) {
  zero_kernel<<<(int)((n + 255) / 256), 256, 0, s>>>(p, n);
}

// Pack weights then run the WMMA GEMM. Abf must already be padded bf16 [M,Kp].
static inline void gemm_full(const unsigned short* Abf, const float* B,
                             const float* bias, float* C,
                             int M, int N, int K, int Kp,
                             unsigned short* BPK, hipStream_t s)
{
  const int ntiles4 = (((N + 15) / 16 + 3) / 4) * 4;   // padded to multiple of 4
  const int kchunks = Kp / 32;
  pack_B_frag<<<kchunks * ntiles4, 32, 0, s>>>(B, BPK, K, N, ntiles4);
  const int mtiles = M / 16;          // all M here are multiples of 16
  dim3 grid(ntiles4 / 4, (mtiles + 3) / 4);
  gemm_wmma<<<grid, 128, 0, s>>>(Abf, BPK, bias, C, N, Kp, ntiles4, mtiles);
}

static inline void launch_cvt(const float* X, unsigned short* Y, long M, int K, int Kp,
                              hipStream_t s) {
  long total = M * Kp;
  to_bf16_pad<<<(int)((total + 255) / 256), 256, 0, s>>>(X, Y, M, K, Kp);
}

extern "C" void kernel_launch(void* const* d_in, const int* in_sizes, int n_in,
                              void* d_out, int out_size, void* d_ws, size_t ws_size,
                              hipStream_t stream)
{
  const float* atom_features = (const float*)d_in[0];
  const float* r_vec    = (const float*)d_in[1];
  const float* angle_h  = (const float*)d_in[2];
  const int*   g_src    = (const int*)d_in[3];
  const int*   g_dst    = (const int*)d_in[4];
  const int*   lg_src   = (const int*)d_in[5];
  const int*   lg_dst   = (const int*)d_in[6];
  const int*   node_gid = (const int*)d_in[7];
  const float* emb_W    = (const float*)d_in[8];
  const float* emb_b    = (const float*)d_in[9];
  const float* c1_Wsrc  = (const float*)d_in[10];
  const float* c1_bsrc  = (const float*)d_in[11];
  const float* c1_Wdst  = (const float*)d_in[12];
  const float* c1_bdst  = (const float*)d_in[13];
  const float* c1_Wedge = (const float*)d_in[14];
  const float* c1_bedge = (const float*)d_in[15];
  const float* c1_bnm_g = (const float*)d_in[16];
  const float* c1_bnm_b = (const float*)d_in[17];
  const float* c1_bn_g  = (const float*)d_in[18];
  const float* c1_bn_b  = (const float*)d_in[19];
  const float* c2_Wsrc  = (const float*)d_in[20];
  const float* c2_bsrc  = (const float*)d_in[21];
  const float* c2_Wdst  = (const float*)d_in[22];
  const float* c2_bdst  = (const float*)d_in[23];
  const float* c2_Wedge = (const float*)d_in[24];
  const float* c2_bedge = (const float*)d_in[25];
  const float* c2_bnm_g = (const float*)d_in[26];
  const float* c2_bnm_b = (const float*)d_in[27];
  const float* c2_bn_g  = (const float*)d_in[28];
  const float* c2_bn_b  = (const float*)d_in[29];
  const float* fc_W     = (const float*)d_in[30];
  const float* fc_b     = (const float*)d_in[31];
  const float* out_W    = (const float*)d_in[32];
  const float* out_b    = (const float*)d_in[33];
  float* out = (float*)d_out;
  float* ws  = (float*)d_ws;

  // Padded K dims (multiple of 32)
  const int KP_NF = 96;   // 92 -> 96
  const int KP_EF = 64;   // 40 -> 64

  // --- workspace layout (float units, 1KB-aligned chunks) ---
  size_t off = 0;
  auto take = [&](size_t nfloats) {
    size_t r = off; off += (nfloats + 255) & ~(size_t)255; return r;
  };
  float* EFb  = ws + take((size_t)NE_ * EFD_);       // edge RBF / line-node feats
  float* AFb  = ws + take((size_t)NA_ * EFD_);       // angle RBF
  float* NFb  = ws + take((size_t)NN_ * NFD_);       // node feats
  float* HSRC = ws + take((size_t)NE_ * 2 * EFD_);   // max(NN*184, NE*80)
  float* HDST = ws + take((size_t)NE_ * 2 * EFD_);
  float* MB   = ws + take((size_t)NE_ * 2 * NFD_);   // max(NE*184, NA*80)
  float* HACC = ws + take((size_t)NE_ * EFD_);       // max(NN*92, NE*40)
  float* BL   = ws + take((size_t)NE_);
  float* CS   = ws + take(256);
  float* CSS  = ws + take(256);
  float* SC   = ws + take(256);
  float* SH   = ws + take(256);
  float* PS   = ws + take((size_t)NG_ * NFD_);
  float* PC   = ws + take(256);
  float* FT   = ws + take((size_t)NG_ * NFD_);
  float* FCb  = ws + take((size_t)NG_ * FCD_);
  // bf16 staging (ushort) — sizes given in float units (2 ushorts / float)
  unsigned short* ABF_N = (unsigned short*)(ws + take(((size_t)NN_ * KP_NF + 1) / 2));
  unsigned short* ABF_E = (unsigned short*)(ws + take(((size_t)NE_ * KP_EF + 1) / 2));
  unsigned short* ABF_A = (unsigned short*)(ws + take(((size_t)NA_ * KP_EF + 1) / 2));
  unsigned short* BPK   = (unsigned short*)(ws + take(32768));  // packed weights
  (void)ws_size; (void)in_sizes; (void)n_in; (void)out_size;

  const float PI_ = 3.14159265358979323846f;

  // --- featurization ---
  bondlen_kernel<<<(NE_ + 255) / 256, 256, 0, stream>>>(r_vec, BL, NE_);
  rbf_kernel<<<(int)(((long)NE_ * EFD_ + 255) / 256), 256, 0, stream>>>(
      BL, EFb, NE_, 0.f, 8.f / 39.f, EFD_);
  rbf_kernel<<<(int)(((long)NA_ * EFD_ + 255) / 256), 256, 0, stream>>>(
      angle_h, AFb, NA_, -PI_ * 0.5f, PI_ / 39.f, EFD_);
  launch_cvt(AFb, ABF_A, NA_, EFD_, KP_EF, stream);   // angles never change
  launch_cvt(EFb, ABF_E, NE_, EFD_, KP_EF, stream);

  // --- embedding: nf = atom_features @ emb_W + emb_b ---
  launch_cvt(atom_features, ABF_N, NN_, NFD_, KP_NF, stream);
  gemm_full(ABF_N, emb_W, emb_b, NFb, NN_, NFD_, NFD_, KP_NF, BPK, stream);
  launch_cvt(NFb, ABF_N, NN_, NFD_, KP_NF, stream);

  for (int i = 0; i < LAYERS_; ++i) {
    // ================= conv1: atom graph (Fin=92, Fo=184) =================
    const int Fo1 = 2 * NFD_;
    gemm_full(ABF_N, c1_Wsrc + (size_t)i * NFD_ * Fo1, c1_bsrc + (size_t)i * Fo1,
              HSRC, NN_, Fo1, NFD_, KP_NF, BPK, stream);
    gemm_full(ABF_N, c1_Wdst + (size_t)i * NFD_ * Fo1, c1_bdst + (size_t)i * Fo1,
              HDST, NN_, Fo1, NFD_, KP_NF, BPK, stream);
    gemm_full(ABF_E, c1_Wedge + (size_t)i * EFD_ * Fo1, c1_bedge + (size_t)i * Fo1,
              MB, NE_, Fo1, EFD_, KP_EF, BPK, stream);
    launch_zero(CS, 256, stream);
    launch_zero(CSS, 256, stream);
    gather_msg_stats<<<(NE_ + 255) / 256, 192, 0, stream>>>(
        HSRC, HDST, MB, g_src, g_dst, CS, CSS, NE_, Fo1, 256);
    bn_finalize<<<1, 256, 0, stream>>>(CS, CSS, c1_bnm_g + (size_t)i * Fo1,
                                       c1_bnm_b + (size_t)i * Fo1, SC, SH,
                                       Fo1, 1.f / NE_);
    launch_zero(HACC, (long)NN_ * NFD_, stream);
    {
      long total = (long)NE_ * NFD_;
      gate_scatter<<<(int)((total + 255) / 256), 256, 0, stream>>>(
          MB, SC, SH, g_dst, HACC, total, NFD_);
    }
    launch_zero(CS, 256, stream);
    launch_zero(CSS, 256, stream);
    col_stats<<<(NN_ + 255) / 256, 96, 0, stream>>>(HACC, CS, CSS, NN_, NFD_, 256);
    bn_finalize<<<1, 256, 0, stream>>>(CS, CSS, c1_bn_g + (size_t)i * NFD_,
                                       c1_bn_b + (size_t)i * NFD_, SC, SH,
                                       NFD_, 1.f / NN_);
    {
      long total = (long)NN_ * NFD_;
      node_update<<<(int)((total + 255) / 256), 256, 0, stream>>>(
          NFb, HACC, SC, SH, total, NFD_);
    }
    launch_cvt(NFb, ABF_N, NN_, NFD_, KP_NF, stream);

    // ================= conv2: line graph (Fin=40, Fo=80) =================
    const int Fo2 = 2 * EFD_;
    gemm_full(ABF_E, c2_Wsrc + (size_t)i * EFD_ * Fo2, c2_bsrc + (size_t)i * Fo2,
              HSRC, NE_, Fo2, EFD_, KP_EF, BPK, stream);
    gemm_full(ABF_E, c2_Wdst + (size_t)i * EFD_ * Fo2, c2_bdst + (size_t)i * Fo2,
              HDST, NE_, Fo2, EFD_, KP_EF, BPK, stream);
    gemm_full(ABF_A, c2_Wedge + (size_t)i * EFD_ * Fo2, c2_bedge + (size_t)i * Fo2,
              MB, NA_, Fo2, EFD_, KP_EF, BPK, stream);
    launch_zero(CS, 256, stream);
    launch_zero(CSS, 256, stream);
    gather_msg_stats<<<(NA_ + 255) / 256, 96, 0, stream>>>(
        HSRC, HDST, MB, lg_src, lg_dst, CS, CSS, NA_, Fo2, 256);
    bn_finalize<<<1, 256, 0, stream>>>(CS, CSS, c2_bnm_g + (size_t)i * Fo2,
                                       c2_bnm_b + (size_t)i * Fo2, SC, SH,
                                       Fo2, 1.f / NA_);
    launch_zero(HACC, (long)NE_ * EFD_, stream);
    {
      long total = (long)NA_ * EFD_;
      gate_scatter<<<(int)((total + 255) / 256), 256, 0, stream>>>(
          MB, SC, SH, lg_dst, HACC, total, EFD_);
    }
    launch_zero(CS, 256, stream);
    launch_zero(CSS, 256, stream);
    col_stats<<<(NE_ + 255) / 256, 64, 0, stream>>>(HACC, CS, CSS, NE_, EFD_, 256);
    bn_finalize<<<1, 256, 0, stream>>>(CS, CSS, c2_bn_g + (size_t)i * EFD_,
                                       c2_bn_b + (size_t)i * EFD_, SC, SH,
                                       EFD_, 1.f / NE_);
    {
      long total = (long)NE_ * EFD_;
      node_update<<<(int)((total + 255) / 256), 256, 0, stream>>>(
          EFb, HACC, SC, SH, total, EFD_);
    }
    launch_cvt(EFb, ABF_E, NE_, EFD_, KP_EF, stream);
  }

  // ================= readout =================
  launch_zero(PS, (long)NG_ * NFD_, stream);
  launch_zero(PC, NG_, stream);
  {
    long total = (long)NN_ * NFD_;
    pool_kernel<<<(int)((total + 255) / 256), 256, 0, stream>>>(
        NFb, node_gid, PS, PC, total, NFD_);
  }
  {
    long total = (long)NG_ * NFD_;
    pool_finalize<<<(int)((total + 255) / 256), 256, 0, stream>>>(
        FT, PS, PC, total, NFD_);
  }
  launch_cvt(FT, ABF_N, NG_, NFD_, KP_NF, stream);     // reuse node staging buffer
  gemm_full(ABF_N, fc_W, fc_b, FCb, NG_, FCD_, NFD_, KP_NF, BPK, stream);
  {
    long total = (long)NG_ * FCD_;
    softplus2_kernel<<<(int)((total + 255) / 256), 256, 0, stream>>>(FCb, total);
  }
  out_kernel<<<1, 256, 0, stream>>>(FCb, out_W, out_b, out, NG_, FCD_);
}
